// mLSTMblock_37735582663290
// MI455X (gfx1250) — matrix operand order repaired and checked by reading, not compiled, not run on hardware
//
#include <hip/hip_runtime.h>
#include <math.h>

// ---------------------------------------------------------------------------
// mLSTM block for MI455X (gfx1250, wave32, WMMA, 320KB-LDS WGP).
//
// Decomposition:
//   Wx[b,t,g]    = x @ W_w^T + W_b                  (WMMA bf16 GEMM, 4.3 GF)
//   A[t,b,o,h]   = sum_d x[b,t,d] * B_w[o,d,h]      (WMMA bf16 GEMM, 275 GF)
//   scan over t:  gates = Wx_t + h U_w^T + U_b ; m = sum_h A_t[o,h] h[h]
//                 c = f*c + i*g + 0.1*tanh(m+B_b) ; h = o*tanh(c)
//
// Constants: B=16 T=512 D=256 H=256, 4H=1024, N_A = O*H = 65536, M = B*T = 8192.
// ---------------------------------------------------------------------------

typedef __attribute__((ext_vector_type(16))) __bf16 v16bf;
typedef __attribute__((ext_vector_type(8)))  float  v8f;

__device__ __forceinline__ __bf16 f32_to_bf16(float f) {
  union { float f; unsigned u; } in; in.f = f;
  unsigned u = in.u;
  u += 0x7FFFu + ((u >> 16) & 1u);          // round-to-nearest-even
  union { unsigned short s; __bf16 b; } out; out.s = (unsigned short)(u >> 16);
  return out.b;
}
__device__ __forceinline__ float bf16_to_f32(__bf16 b) {
  union { unsigned short s; __bf16 b; } in; in.b = b;
  union { unsigned u; float f; } out; out.u = ((unsigned)in.s) << 16;
  return out.f;
}
__device__ __forceinline__ float sigmoidf_(float x) { return 1.0f / (1.0f + __expf(-x)); }

// ---------------------------------------------------------------------------
// Pack kernels: gather f32 sources into wave32 WMMA bf16 fragment layouts.
// A-fragment (16x32, MxK): lane l=(m,half): 16 elems = K in
//   [half*16, half*16+16)
// B-fragment (32x16, KxN): lane l=(n,half): 16 elems = K in [half*16, half*16+16)
// Fragment storage: [tile][lane][16] contiguous -> one 32B load per lane.
// ---------------------------------------------------------------------------

// x [8192,256] -> xpack, tiles = mt(512) x kt(8), tile index = mt*8+kt
__global__ void pack_x_kernel(const float* __restrict__ x, __bf16* __restrict__ xpack) {
  int wave = threadIdx.x >> 5, lane = threadIdx.x & 31;
  int tile = blockIdx.x * 8 + wave;                 // 4096 tiles
  int mt = tile >> 3, kt = tile & 7;
  int m_local = lane & 15, half = lane >> 4;
  int row = mt * 16 + m_local;
  __bf16* dst = xpack + ((size_t)tile * 32 + lane) * 16;
  int kbase = kt * 32 + half * 16;
#pragma unroll
  for (int e = 0; e < 8; ++e)  dst[e]     = f32_to_bf16(x[row * 256 + kbase + e]);
#pragma unroll
  for (int e = 0; e < 8; ++e)  dst[8 + e] = f32_to_bf16(x[row * 256 + kbase + 8 + e]);
}

// B_w [O=256, D=256, H=256]; GEMM B matrix W[d, n] with n = o*256+h.
// tiles = kt(8) x nt(4096), tile index = kt*4096+nt
__global__ void pack_Bw_kernel(const float* __restrict__ Bw, __bf16* __restrict__ bpack) {
  int wave = threadIdx.x >> 5, lane = threadIdx.x & 31;
  int tile = blockIdx.x * 8 + wave;                 // 32768 tiles
  int kt = tile >> 12, nt = tile & 4095;
  int n_local = lane & 15, half = lane >> 4;
  int n = nt * 16 + n_local;
  int o = n >> 8, hh = n & 255;
  __bf16* dst = bpack + ((size_t)tile * 32 + lane) * 16;
  int dbase = kt * 32 + half * 16;
#pragma unroll
  for (int e = 0; e < 16; ++e)
    dst[e] = f32_to_bf16(Bw[((size_t)o << 16) + (size_t)(dbase + e) * 256 + hh]);
}

// W_w [1024,256]; GEMM B matrix W[d, g] = W_w[g, d]. tiles = kt(8) x nt(64)
__global__ void pack_Ww_kernel(const float* __restrict__ Ww, __bf16* __restrict__ wpack) {
  int wave = threadIdx.x >> 5, lane = threadIdx.x & 31;
  int tile = blockIdx.x * 8 + wave;                 // 512 tiles
  int kt = tile >> 6, nt = tile & 63;
  int n_local = lane & 15, half = lane >> 4;
  int n = nt * 16 + n_local;
  __bf16* dst = wpack + ((size_t)tile * 32 + lane) * 16;
  int dbase = kt * 32 + half * 16;
#pragma unroll
  for (int e = 0; e < 16; ++e)
    dst[e] = f32_to_bf16(Ww[n * 256 + dbase + e]);
}

// ---------------------------------------------------------------------------
// WMMA GEMM: Wx[m, n] = sum_d x[m,d] W_w[n,d] + W_b[n]   (M=8192,N=1024,K=256)
// Each wave: 16x64 tile (4 accumulators, A-fragment reuse). 8 waves/block.
// ---------------------------------------------------------------------------
__global__ void gemm_wx_kernel(const __bf16* __restrict__ xpack,
                               const __bf16* __restrict__ wpack,
                               const float*  __restrict__ Wb,
                               float* __restrict__ Wx) {
  int wave = threadIdx.x >> 5, lane = threadIdx.x & 31;
  int mt = blockIdx.x >> 1;                                // 512 M tiles
  int nGroup = (blockIdx.x & 1) * 8 + wave;                // 16 groups of 64 cols
  v8f acc[4] = {};
#pragma unroll
  for (int kt = 0; kt < 8; ++kt) {
    v16bf a = *(const v16bf*)(xpack + ((size_t)(mt * 8 + kt) * 32 + lane) * 16);
#pragma unroll
    for (int j = 0; j < 4; ++j) {
      int nt = nGroup * 4 + j;
      v16bf b = *(const v16bf*)(wpack + ((size_t)(kt * 64 + nt) * 32 + lane) * 16);
      acc[j] = __builtin_amdgcn_wmma_f32_16x16x32_bf16(
          false, a, false, b, (short)0, acc[j], false, false);
    }
  }
  int col = lane & 15, rhalf = lane >> 4;
#pragma unroll
  for (int j = 0; j < 4; ++j) {
    int n = (nGroup * 4 + j) * 16 + col;
    float bias = Wb[n];
#pragma unroll
    for (int r = 0; r < 8; ++r) {
      int m = mt * 16 + rhalf * 8 + r;
      Wx[(size_t)m * 1024 + n] = acc[j][r] + bias;
    }
  }
}

// ---------------------------------------------------------------------------
// WMMA GEMM: A[t,b,n] = sum_d x[(b,t),d] * Bw_packed[d,n]  (M=8192,N=65536,K=256)
// Output stored bf16 as A[(t*16+b)*65536 + n] so the scan reads contiguous rows.
// ---------------------------------------------------------------------------
__global__ void gemm_A_kernel(const __bf16* __restrict__ xpack,
                              const __bf16* __restrict__ bpack,
                              __bf16* __restrict__ Abuf) {
  int wave = threadIdx.x >> 5, lane = threadIdx.x & 31;
  int mt = blockIdx.x >> 7;                                // 512 M tiles
  int nGroup = (blockIdx.x & 127) * 8 + wave;              // 1024 groups of 64 cols
  v8f acc[4] = {};
#pragma unroll
  for (int kt = 0; kt < 8; ++kt) {
    v16bf a = *(const v16bf*)(xpack + ((size_t)(mt * 8 + kt) * 32 + lane) * 16);
#pragma unroll
    for (int j = 0; j < 4; ++j) {
      int nt = nGroup * 4 + j;
      v16bf b = *(const v16bf*)(bpack + ((size_t)(kt * 4096 + nt) * 32 + lane) * 16);
      acc[j] = __builtin_amdgcn_wmma_f32_16x16x32_bf16(
          false, a, false, b, (short)0, acc[j], false, false);
    }
  }
  int col = lane & 15, rhalf = lane >> 4;
#pragma unroll
  for (int j = 0; j < 4; ++j) {
    int n = (nGroup * 4 + j) * 16 + col;
#pragma unroll
    for (int r = 0; r < 8; ++r) {
      int m = mt * 16 + rhalf * 8 + r;       // m = b*512 + t
      int bb = m >> 9, tt = m & 511;
      Abuf[(((size_t)(tt * 16 + bb)) << 16) + n] = f32_to_bf16(acc[j][r]);
    }
  }
}

// ---------------------------------------------------------------------------
// Persistent scan kernel, 64 blocks x 256 threads (wave32: 8 waves/block).
// Block blk serves batch b = blk>>2:
//   gate thread: g = (blk&3)*256 + tid  -> 256-FMA dot out of LDS
//   m-partial  : o = (blk&3)*64 + tid/4, quarter q = tid&3 -> 64-FMA dot (A from HBM)
// Blocks 0..15 additionally own the state update for (b2=blk, o2=tid).
// LDS (264KB of the 320KB WGP LDS): 256 U_w rows (stride 257 -> conflict-free
// banks) + current h row.
// ---------------------------------------------------------------------------
__device__ __forceinline__ void grid_barrier(unsigned* cnt, unsigned* gen, unsigned nBlocks) {
  __threadfence();
  __syncthreads();
  if (threadIdx.x == 0) {
    unsigned g = __hip_atomic_load(gen, __ATOMIC_RELAXED, __HIP_MEMORY_SCOPE_AGENT);
    unsigned arrived =
        __hip_atomic_fetch_add(cnt, 1u, __ATOMIC_ACQ_REL, __HIP_MEMORY_SCOPE_AGENT);
    if (arrived == nBlocks - 1) {
      __hip_atomic_store(cnt, 0u, __ATOMIC_RELAXED, __HIP_MEMORY_SCOPE_AGENT);
      __hip_atomic_fetch_add(gen, 1u, __ATOMIC_RELEASE, __HIP_MEMORY_SCOPE_AGENT);
    } else {
      while (__hip_atomic_load(gen, __ATOMIC_ACQUIRE, __HIP_MEMORY_SCOPE_AGENT) == g)
        __builtin_amdgcn_s_sleep(1);
    }
  }
  __syncthreads();
  __threadfence();
}

__global__ void init_kernel(float* hG, unsigned* bar) {
  for (int i = threadIdx.x; i < 4096; i += blockDim.x) hG[i] = 0.0f;
  if (threadIdx.x < 2) bar[threadIdx.x] = 0u;
}

#define U_STRIDE 257   // 256 + 1: bank = (row*257 + h) % 64 unique per lane

__global__ void scan_kernel(const float* __restrict__ Wx,
                            const __bf16* __restrict__ Abuf,
                            const float* __restrict__ Uw,
                            const float* __restrict__ Ub,
                            const float* __restrict__ Bb,
                            float* __restrict__ hG,
                            float* __restrict__ gatesG,
                            float* __restrict__ mPart,
                            unsigned* bar,
                            float* __restrict__ out) {
  extern __shared__ char smem[];
  float* u_lds = (float*)smem;                            // 256 * 257 floats
  float* h_lds = (float*)(smem + 256 * U_STRIDE * 4);     // 256 floats

  const int tid = threadIdx.x;
  const int blk = blockIdx.x;
  const int gt  = blk * 256 + tid;                        // 0..16383
  const int b   = blk >> 2;                               // batch for this block
  const int g   = ((blk & 3) << 8) + tid;                 // gate row 0..1023
  const int oq  = ((blk & 3) << 6) + (tid >> 2);          // o for m-partial
  const int q   = tid & 3;                                // quarter of h-dot
  const unsigned nBlocks = gridDim.x;

  // One-time: stage this block's 256 U_w rows into LDS.
  for (int k = 0; k < 256; ++k) u_lds[tid * U_STRIDE + k] = Uw[g * 256 + k];
  __syncthreads();

  float creg = 0.0f, hlast = 0.0f;

  for (int t = 0; t < 512; ++t) {
    // stage h_{t-1}[b] into LDS (1KB broadcast row)
    h_lds[tid] = hG[b * 256 + tid];
    __syncthreads();

    // --- phase A1: gate pre-activation, entirely out of LDS ---
    float acc = Wx[((size_t)(b * 512 + t)) * 1024 + g] + Ub[g];
    const float* urow = u_lds + tid * U_STRIDE;
#pragma unroll 4
    for (int hh = 0; hh < 256; ++hh) acc = fmaf(h_lds[hh], urow[hh], acc);
    gatesG[gt] = acc;

    // --- phase A2: bilinear partial: 64-length dot of A[t,b,oq,:] with h ---
    const __bf16* arow = Abuf + (((size_t)(t * 16 + b)) << 16) + (oq << 8) + (q << 6);
    if (t < 511)  // next timestep's 128B chunk for this thread
      __builtin_prefetch((const void*)(arow + ((size_t)16 << 16)), 0, 1);
    float mp = 0.0f;
#pragma unroll
    for (int j = 0; j < 4; ++j) {
      v16bf av = *(const v16bf*)(arow + j * 16);
      const float* hseg = h_lds + (q << 6) + (j << 4);
#pragma unroll
      for (int k = 0; k < 16; ++k) mp = fmaf(bf16_to_f32(av[k]), hseg[k], mp);
    }
    mPart[gt] = mp;

    grid_barrier(bar, bar + 1, nBlocks);

    // --- phase B: state update (blocks 0..15: b2 = blk, o2 = tid) ---
    if (blk < 16) {
      const float* gb = gatesG + blk * 1024;
      float iv = sigmoidf_(gb[tid]);
      float fv = sigmoidf_(gb[256 + tid]);
      float ov = sigmoidf_(gb[512 + tid]);
      float gv = tanhf(gb[768 + tid]);
      int e4 = 4 * (blk * 256 + tid);
      float msum = Bb[tid] + mPart[e4] + mPart[e4 + 1] + mPart[e4 + 2] + mPart[e4 + 3];
      float mv = tanhf(msum);
      creg = fv * creg + iv * gv + 0.1f * mv;
      hlast = ov * tanhf(creg);
      hG[blk * 256 + tid] = hlast;
      out[((size_t)(blk * 512 + t)) * 256 + tid] = hlast;   // outs [B,T,H]
    }
    grid_barrier(bar, bar + 1, nBlocks);
  }

  if (blk < 16) {                                           // final (h, c)
    out[2097152 + blk * 256 + tid] = hlast;
    out[2097152 + 4096 + blk * 256 + tid] = creg;
  }
}

// ---------------------------------------------------------------------------
extern "C" void kernel_launch(void* const* d_in, const int* in_sizes, int n_in,
                              void* d_out, int out_size, void* d_ws, size_t ws_size,
                              hipStream_t stream) {
  const float* x   = (const float*)d_in[0];
  const float* W_w = (const float*)d_in[1];
  const float* W_b = (const float*)d_in[2];
  const float* U_w = (const float*)d_in[3];
  const float* U_b = (const float*)d_in[4];
  const float* B_w = (const float*)d_in[5];
  const float* B_b = (const float*)d_in[6];
  float* out = (float*)d_out;

  char* ws = (char*)d_ws;
  size_t off = 0;
  auto carve = [&](size_t bytes) -> void* {
    off = (off + 255) & ~(size_t)255;
    void* p = ws + off;
    off += bytes;
    return p;
  };
  float*    Wx     = (float*)   carve((size_t)8192 * 1024 * 4);        // 33.5 MB
  float*    gatesG = (float*)   carve((size_t)16384 * 4);
  float*    mPart  = (float*)   carve((size_t)16384 * 4);
  float*    hG     = (float*)   carve((size_t)4096 * 4);
  unsigned* bar    = (unsigned*)carve(256);
  __bf16*   xpack  = (__bf16*)  carve((size_t)4096 * 32 * 16 * 2);     // 4.2 MB
  __bf16*   wpack  = (__bf16*)  carve((size_t)512 * 32 * 16 * 2);      // 0.5 MB
  __bf16*   bpack  = (__bf16*)  carve((size_t)32768 * 32 * 16 * 2);    // 33.5 MB
  __bf16*   Abuf   = (__bf16*)  carve((size_t)512 * 16 * 65536 * 2);   // 1.07 GB
  if (off > ws_size) return;  // workspace too small; nothing safe to do

  const size_t scan_lds = (size_t)(256 * U_STRIDE + 256) * 4;          // ~264 KB

  init_kernel<<<1, 256, 0, stream>>>(hG, bar);
  pack_x_kernel <<<512,   256, 0, stream>>>(x, xpack);
  pack_Ww_kernel<<<64,    256, 0, stream>>>(W_w, wpack);
  pack_Bw_kernel<<<4096,  256, 0, stream>>>(B_w, bpack);
  gemm_wx_kernel<<<1024,  256, 0, stream>>>(xpack, wpack, W_b, Wx);
  gemm_A_kernel <<<65536, 256, 0, stream>>>(xpack, bpack, Abuf);
  scan_kernel   <<<64, 256, scan_lds, stream>>>(Wx, Abuf, U_w, U_b, B_b,
                                                hG, gatesG, mPart, bar, out);
}
